// SelfAttentionBlock_47502338293815
// MI455X (gfx1250) — matrix operand order repaired and checked by reading, not compiled, not run on hardware
//
#include <hip/hip_runtime.h>

typedef __bf16 bf16_t;
typedef __attribute__((ext_vector_type(16))) __bf16 v16bf;
typedef __attribute__((ext_vector_type(8)))  __bf16 v8bf;
typedef __attribute__((ext_vector_type(8)))  float  v8f;

static constexpr int kB   = 16;
static constexpr int kCQ  = 512;
static constexpr int kCT  = 256;
static constexpr int kCO  = 512;
static constexpr int kL   = 1024;   // 32*32
static constexpr int kNB  = 3969;
static constexpr int kNBP = 3984;   // 249 * 16 (padded)

// ---------------- WMMA helpers (CDNA5 bf16 16x16x32, f32 accum) ----------------

__device__ __forceinline__ v8f wmma_bf16(v16bf a, v16bf b, v8f c) {
  // (neg_a, A, neg_b, B, c_mod, C, reuse_a, reuse_b)
  return __builtin_amdgcn_wmma_f32_16x16x32_bf16(false, a, false, b, (short)0, c,
                                                 false, false);
}

// A-matrix 16x32 bf16 fragment from row-major [rows, ld]:
// lane m=lane&15 holds row (row0+m); lanes 0-15 hold K {0..7,16..23}, lanes 16-31 K {8..15,24..31}.
__device__ __forceinline__ v16bf load_frag_a(const bf16_t* p, int row0, int k0, int ld) {
  int lane = threadIdx.x & 31;
  int m = lane & 15, hi = lane >> 4;
  const bf16_t* r = p + (size_t)(row0 + m) * ld + k0 + hi * 8;
  v8bf lo = *(const v8bf*)r;
  v8bf h8 = *(const v8bf*)(r + 16);
  v16bf f;
#pragma unroll
  for (int i = 0; i < 8; ++i) { f[i] = lo[i]; f[8 + i] = h8[i]; }
  return f;
}

// B-matrix 32x16 fragment where logical B[K][N] = src[(col0+N)*ld + k0 + K]
// (i.e. src holds the N-dimension as rows -> contiguous per-lane loads).
// lane n=lane&15 holds column (col0+n); lanes 0-15 K {0..7,16..23}, lanes 16-31 K {8..15,24..31}.
__device__ __forceinline__ v16bf load_frag_b(const bf16_t* p, int col0, int k0, int ld) {
  int lane = threadIdx.x & 31;
  int n = lane & 15, hi = lane >> 4;
  const bf16_t* r = p + (size_t)(col0 + n) * ld + k0 + hi * 8;
  v8bf lo = *(const v8bf*)r;
  v8bf h8 = *(const v8bf*)(r + 16);
  v16bf f;
#pragma unroll
  for (int i = 0; i < 8; ++i) { f[i] = lo[i]; f[8 + i] = h8[i]; }
  return f;
}

// A fragment built from an f32 matrix in LDS (row stride ldp floats)
__device__ __forceinline__ v16bf frag_a_from_f32(const float* P, int k0, int ldp) {
  int lane = threadIdx.x & 31;
  int m = lane & 15, b0 = (lane >> 4) * 8;
  const float* r = P + (size_t)m * ldp + k0 + b0;
  v16bf f;
#pragma unroll
  for (int i = 0; i < 8; ++i) { f[i] = (bf16_t)r[i]; f[8 + i] = (bf16_t)r[16 + i]; }
  return f;
}

// One 16-wide output tile over K=256: preload all 8 B fragments, then 8 WMMAs.
// (loads issue as one burst -> partial s_wait_loadcnt overlap with the WMMA chain)
__device__ __forceinline__ v8f tile_k256(const v16bf aq[8], const bf16_t* p, int col0, int ld) {
  v16bf bf[8];
#pragma unroll
  for (int kc = 0; kc < 8; ++kc) bf[kc] = load_frag_b(p, col0, kc * 32, ld);
  v8f acc;
#pragma unroll
  for (int r = 0; r < 8; ++r) acc[r] = 0.0f;
#pragma unroll
  for (int kc = 0; kc < 8; ++kc) acc = wmma_bf16(aq[kc], bf[kc], acc);
  return acc;
}

// ---------------- small utility kernels ----------------

__global__ void f32_to_bf16(const float* __restrict__ in, bf16_t* __restrict__ out, int n) {
  int i = blockIdx.x * 256 + threadIdx.x;
  if (i < n) out[i] = (bf16_t)in[i];
}

// lut_w [CT, NB] row-major  ->  lutT [NBP, CT] bf16 (transposed, padded with zeros)
__global__ void lut_repack(const float* __restrict__ lut, bf16_t* __restrict__ lutT) {
  int i = blockIdx.x * 256 + threadIdx.x;   // over kNBP*kCT
  int nb = i / kCT, c = i % kCT;
  lutT[i] = (nb < kNB) ? (bf16_t)lut[(size_t)c * kNB + nb] : (bf16_t)0.0f;
}

// X [B, C, L] f32 (optionally scaled by att[B,C]) -> Xout [B, L, C] bf16
// 32x32 LDS tile transpose: coalesced reads AND writes.
__global__ void convert_nchw_to_nlc(const float* __restrict__ Xin, const float* __restrict__ att,
                                    bf16_t* __restrict__ Xout, int C) {
  __shared__ float tile[32][33];
  int b = blockIdx.z;
  int c0 = blockIdx.y * 32, l0 = blockIdx.x * 32;
  int tx = threadIdx.x, ty = threadIdx.y;   // block (32, 8)
#pragma unroll
  for (int j = 0; j < 4; ++j) {
    int c = c0 + ty + j * 8;
    tile[ty + j * 8][tx] = Xin[((size_t)b * C + c) * kL + l0 + tx];
  }
  __syncthreads();
  float a = (att != nullptr) ? att[b * C + c0 + tx] : 1.0f;
#pragma unroll
  for (int j = 0; j < 4; ++j) {
    int l = l0 + ty + j * 8;
    Xout[((size_t)b * kL + l) * C + c0 + tx] = (bf16_t)(a * tile[tx][ty + j * 8]);
  }
}

// ---------------- channel attention (CBAM) ----------------

__global__ void channel_att(const float* __restrict__ qfeats, const float* __restrict__ w1,
                            const float* __restrict__ w2, float* __restrict__ att) {
  __shared__ float avg[kCQ], mxs[kCQ], h[64];
  int b = blockIdx.x;
  int wave = threadIdx.x >> 5, lane = threadIdx.x & 31;
  for (int c = wave; c < kCQ; c += 8) {
    const float* p = qfeats + ((size_t)b * kCQ + c) * kL;
    float s = 0.0f, m = -3.0e38f;
    for (int i = lane; i < kL; i += 32) { float v = p[i]; s += v; m = fmaxf(m, v); }
#pragma unroll
    for (int off = 16; off >= 1; off >>= 1) {
      s += __shfl_xor(s, off, 32);
      m = fmaxf(m, __shfl_xor(m, off, 32));
    }
    if (lane == 0) { avg[c] = s * (1.0f / (float)kL); mxs[c] = m; }
  }
  __syncthreads();
  if (threadIdx.x < 64) {
    int r = threadIdx.x & 31;
    const float* z = (threadIdx.x >= 32) ? mxs : avg;
    float acc = 0.0f;
    for (int c = 0; c < kCQ; ++c) acc += w1[r * kCQ + c] * z[c];
    h[threadIdx.x] = fmaxf(acc, 0.0f);
  }
  __syncthreads();
  for (int o = threadIdx.x; o < kCQ; o += 256) {
    float acc = 0.0f;
#pragma unroll
    for (int r = 0; r < 32; ++r) acc += w2[o * 32 + r] * (h[r] + h[32 + r]);
    att[b * kCQ + o] = 1.0f / (1.0f + __expf(-acc));
  }
}

// ---------------- generic GEMM + BN + ReLU ----------------
// Y[b][co][l] = relu(s[co] * sum_ci W[co][ci] * X[b][l][ci] + t[co])
// Each wave computes a 16(co) x 64(l) strip: one A fragment feeds 4 accumulators.
// layout 0: Y bf16 [B, L, Co]   (row l contiguous over co)
// layout 1: Y bf16 [B, Co, L]
// layout 2: Y f32  [B, Co, L]   (final output)
__global__ void gemm_bn_act(const bf16_t* __restrict__ W, const bf16_t* __restrict__ Xall,
                            const float* __restrict__ s, const float* __restrict__ t,
                            void* __restrict__ Yall, int Co, int Ci, int layout) {
  int b = blockIdx.z;
  int l0 = (blockIdx.x * 8 + (int)(threadIdx.x >> 5)) * 64;
  int co0 = blockIdx.y * 16;
  const bf16_t* X = Xall + (size_t)b * kL * Ci;

  v8f acc[4];
#pragma unroll
  for (int j = 0; j < 4; ++j)
#pragma unroll
    for (int r = 0; r < 8; ++r) acc[j][r] = 0.0f;

  int lane = threadIdx.x & 31, n = lane & 15, hi = lane >> 4;
  for (int k = 0; k < Ci; k += 32) {
    // gfx1250 speculative prefetch (fault-free) of the next k-chunk of X
    __builtin_prefetch(X + (size_t)(l0 + n) * Ci + k + 32, 0, 3);
    __builtin_prefetch(X + (size_t)(l0 + 32 + n) * Ci + k + 32, 0, 3);
    v16bf a = load_frag_a(W, co0, k, Ci);
    v16bf bf[4];
#pragma unroll
    for (int j = 0; j < 4; ++j) bf[j] = load_frag_b(X, l0 + j * 16, k, Ci);
#pragma unroll
    for (int j = 0; j < 4; ++j) acc[j] = wmma_bf16(a, bf[j], acc[j]);
  }

  float sv[8], tv[8];
#pragma unroll
  for (int r = 0; r < 8; ++r) {
    int co = co0 + hi * 8 + r;
    sv[r] = s[co];
    tv[r] = t[co];
  }
#pragma unroll
  for (int j = 0; j < 4; ++j) {
    float y[8];
#pragma unroll
    for (int r = 0; r < 8; ++r) y[r] = fmaxf(sv[r] * acc[j][r] + tv[r], 0.0f);
    int lj = l0 + j * 16;
    if (layout == 0) {
      bf16_t* Y = (bf16_t*)Yall + ((size_t)b * kL + lj + n) * Co + co0 + hi * 8;
      v8bf pk;
#pragma unroll
      for (int r = 0; r < 8; ++r) pk[r] = (bf16_t)y[r];
      *(v8bf*)Y = pk;
    } else if (layout == 1) {
      bf16_t* Y = (bf16_t*)Yall + (size_t)b * Co * kL;
#pragma unroll
      for (int r = 0; r < 8; ++r) Y[(size_t)(co0 + hi * 8 + r) * kL + lj + n] = (bf16_t)y[r];
    } else {
      float* Y = (float*)Yall + (size_t)b * Co * kL;
#pragma unroll
      for (int r = 0; r < 8; ++r) Y[(size_t)(co0 + hi * 8 + r) * kL + lj + n] = y[r];
    }
  }
}

// ---------------- attention megakernel ----------------
// One block per (16-query tile, batch). 8 waves.
// LDS: lt [16][NBP] bf16 (125 KB)  +  scores [16][L] f32 (64 KB)  -> CDNA5 320KB-class LDS.
__global__ void attention_kernel(const bf16_t* __restrict__ qf, const bf16_t* __restrict__ kf,
                                 const bf16_t* __restrict__ vm, const bf16_t* __restrict__ lutT,
                                 const int* __restrict__ rel, bf16_t* __restrict__ ctx) {
  extern __shared__ char smem[];
  bf16_t* lt = (bf16_t*)smem;                          // [16][kNBP]
  float*  sb = (float*)(smem + (size_t)16 * kNBP * 2); // [16][kL]

  int b = blockIdx.y;
  int l0 = blockIdx.x * 16;
  int wave = threadIdx.x >> 5;
  int lane = threadIdx.x & 31;
  int n = lane & 15, hi = lane >> 4;

  const bf16_t* qb = qf + (size_t)b * kL * kCT;   // [L, CT]
  const bf16_t* kb = kf + (size_t)b * kL * kCT;   // [L, CT]
  const bf16_t* vb = vm + (size_t)b * kCT * kL;   // [CT, L]

  // q-tile A fragments (shared by all phases), 8 K-chunks of 32
  v16bf aq[8];
#pragma unroll
  for (int kc = 0; kc < 8; ++kc) aq[kc] = load_frag_a(qb, l0, kc * 32, kCT);

  // phase 1: lt[16, NBP] = qTile . lut  (B-fragments from transposed lutT [NBP, CT])
  for (int j = wave; j < kNBP / 16; j += 8) {
    v8f acc = tile_k256(aq, lutT, j * 16, kCT);
#pragma unroll
    for (int r = 0; r < 8; ++r)
      lt[(size_t)(hi * 8 + r) * kNBP + j * 16 + n] = (bf16_t)acc[r];
  }

  // phase 2: raw scores sb[16, L] = qTile . K^T
  for (int j = wave; j < kL / 16; j += 8) {
    v8f acc = tile_k256(aq, kb, j * 16, kCT);
#pragma unroll
    for (int r = 0; r < 8; ++r)
      sb[(size_t)(hi * 8 + r) * kL + j * 16 + n] = acc[r];
  }
  __syncthreads();

  // gather relative-position bias from LDS and apply scale: (qk + pos) * ct^-0.5
  const float scale = 0.0625f;  // 256^-0.5
  for (int e = threadIdx.x; e < 16 * kL; e += 256) {
    int l = e >> 10, m = e & (kL - 1);
    int bucket = rel[(size_t)(l0 + l) * kL + m];
    sb[e] = (sb[e] + (float)lt[(size_t)l * kNBP + bucket]) * scale;
  }
  __syncthreads();

  // softmax: each wave owns 2 rows, 16 lanes per row
  {
    int row = wave * 2 + hi;
    float* srow = sb + (size_t)row * kL;
    int lc = lane & 15;
    float mx = -3.0e38f;
    for (int i = lc; i < kL; i += 16) mx = fmaxf(mx, srow[i]);
#pragma unroll
    for (int off = 8; off >= 1; off >>= 1) mx = fmaxf(mx, __shfl_xor(mx, off, 16));
    float sum = 0.0f;
    for (int i = lc; i < kL; i += 16) { float p = __expf(srow[i] - mx); srow[i] = p; sum += p; }
#pragma unroll
    for (int off = 8; off >= 1; off >>= 1) sum += __shfl_xor(sum, off, 16);
    float inv = 1.0f / sum;
    for (int i = lc; i < kL; i += 16) srow[i] *= inv;
  }
  __syncthreads();

  // phase 4: ctx[16, CT] = P[16, L] . V[L, CT]
  // One P fragment (from LDS f32) feeds TWO c-tile accumulators -> half the LDS reads.
  {
    bf16_t* cb = ctx + (size_t)b * kL * kCT;      // [L, CT]
    int jc0 = wave * 2, jc1 = wave * 2 + 1;
    v8f acc0, acc1;
#pragma unroll
    for (int r = 0; r < 8; ++r) { acc0[r] = 0.0f; acc1[r] = 0.0f; }
    for (int km = 0; km < kL / 32; ++km) {
      v16bf b0 = load_frag_b(vb, jc0 * 16, km * 32, kL);
      v16bf b1 = load_frag_b(vb, jc1 * 16, km * 32, kL);
      v16bf ap = frag_a_from_f32(sb, km * 32, kL);
      acc0 = wmma_bf16(ap, b0, acc0);
      acc1 = wmma_bf16(ap, b1, acc1);
    }
#pragma unroll
    for (int r = 0; r < 8; ++r) {
      cb[(size_t)(l0 + hi * 8 + r) * kCT + jc0 * 16 + n] = (bf16_t)acc0[r];
      cb[(size_t)(l0 + hi * 8 + r) * kCT + jc1 * 16 + n] = (bf16_t)acc1[r];
    }
  }
}

// ---------------- orchestration ----------------

extern "C" void kernel_launch(void* const* d_in, const int* in_sizes, int n_in,
                              void* d_out, int out_size, void* d_ws, size_t ws_size,
                              hipStream_t stream) {
  (void)in_sizes; (void)n_in; (void)out_size; (void)ws_size;

  const float* query = (const float*)d_in[0];
  const float* key   = (const float*)d_in[1];
  const float* value = (const float*)d_in[2];
  const float* ca_w1 = (const float*)d_in[3];
  const float* ca_w2 = (const float*)d_in[4];
  const float* q_w1 = (const float*)d_in[5];
  const float* q_s1 = (const float*)d_in[6];
  const float* q_t1 = (const float*)d_in[7];
  const float* q_w2 = (const float*)d_in[8];
  const float* q_s2 = (const float*)d_in[9];
  const float* q_t2 = (const float*)d_in[10];
  const float* k_w1 = (const float*)d_in[11];
  const float* k_s1 = (const float*)d_in[12];
  const float* k_t1 = (const float*)d_in[13];
  const float* k_w2 = (const float*)d_in[14];
  const float* k_s2 = (const float*)d_in[15];
  const float* k_t2 = (const float*)d_in[16];
  const float* v_w  = (const float*)d_in[17];
  const float* v_s  = (const float*)d_in[18];
  const float* v_t  = (const float*)d_in[19];
  const float* o_w  = (const float*)d_in[20];
  const float* o_s  = (const float*)d_in[21];
  const float* o_t  = (const float*)d_in[22];
  const float* lut  = (const float*)d_in[23];
  const int*   rel  = (const int*)d_in[24];

  char* ws = (char*)d_ws;
  size_t off = 0;
  auto carve = [&](size_t bytes) -> char* {
    char* p = ws + off;
    off += (bytes + 255) & ~(size_t)255;
    return p;
  };
  float*  att  = (float*)carve((size_t)kB * kCQ * 4);
  bf16_t* wq1  = (bf16_t*)carve((size_t)kCT * kCQ * 2);
  bf16_t* wq2  = (bf16_t*)carve((size_t)kCT * kCT * 2);
  bf16_t* wk1  = (bf16_t*)carve((size_t)kCT * kCQ * 2);
  bf16_t* wk2  = (bf16_t*)carve((size_t)kCT * kCT * 2);
  bf16_t* wv   = (bf16_t*)carve((size_t)kCT * kCQ * 2);
  bf16_t* wo   = (bf16_t*)carve((size_t)kCO * kCT * 2);
  bf16_t* lutT = (bf16_t*)carve((size_t)kNBP * kCT * 2);
  bf16_t* bufA = (bf16_t*)carve((size_t)kB * kL * kCQ * 2);  // 16 MB (xq/xk/xv)
  bf16_t* bufB = (bf16_t*)carve((size_t)kB * kL * kCT * 2);  // 8 MB (stage-1 proj)
  bf16_t* qfb  = (bf16_t*)carve((size_t)kB * kL * kCT * 2);  // q  [B, L, CT]
  bf16_t* kfb  = (bf16_t*)carve((size_t)kB * kL * kCT * 2);  // k  [B, L, CT]
  bf16_t* vmb  = (bf16_t*)carve((size_t)kB * kCT * kL * 2);  // v  [B, CT, L]
  bf16_t* ctx  = (bf16_t*)carve((size_t)kB * kL * kCT * 2);  // ctx [B, L, CT]

  // weight + LUT conversion (bf16, LUT transposed/padded)
  f32_to_bf16<<<512, 256, 0, stream>>>(q_w1, wq1, kCT * kCQ);
  f32_to_bf16<<<256, 256, 0, stream>>>(q_w2, wq2, kCT * kCT);
  f32_to_bf16<<<512, 256, 0, stream>>>(k_w1, wk1, kCT * kCQ);
  f32_to_bf16<<<256, 256, 0, stream>>>(k_w2, wk2, kCT * kCT);
  f32_to_bf16<<<512, 256, 0, stream>>>(v_w,  wv,  kCT * kCQ);
  f32_to_bf16<<<512, 256, 0, stream>>>(o_w,  wo,  kCO * kCT);
  lut_repack<<<kNBP, 256, 0, stream>>>(lut, lutT);  // kNBP*kCT / 256 blocks

  // channel attention
  channel_att<<<kB, 256, 0, stream>>>(query, ca_w1, ca_w2, att);

  // q chain
  convert_nchw_to_nlc<<<dim3(kL / 32, kCQ / 32, kB), dim3(32, 8), 0, stream>>>(query, att, bufA, kCQ);
  gemm_bn_act<<<dim3(2, kCT / 16, kB), 256, 0, stream>>>(wq1, bufA, q_s1, q_t1, bufB, kCT, kCQ, 0);
  gemm_bn_act<<<dim3(2, kCT / 16, kB), 256, 0, stream>>>(wq2, bufB, q_s2, q_t2, qfb, kCT, kCT, 0);
  // k chain
  convert_nchw_to_nlc<<<dim3(kL / 32, kCQ / 32, kB), dim3(32, 8), 0, stream>>>(key, nullptr, bufA, kCQ);
  gemm_bn_act<<<dim3(2, kCT / 16, kB), 256, 0, stream>>>(wk1, bufA, k_s1, k_t1, bufB, kCT, kCQ, 0);
  gemm_bn_act<<<dim3(2, kCT / 16, kB), 256, 0, stream>>>(wk2, bufB, k_s2, k_t2, kfb, kCT, kCT, 0);
  // v (stored [CT, L] for contiguous ctx B-fragments)
  convert_nchw_to_nlc<<<dim3(kL / 32, kCQ / 32, kB), dim3(32, 8), 0, stream>>>(value, nullptr, bufA, kCQ);
  gemm_bn_act<<<dim3(2, kCT / 16, kB), 256, 0, stream>>>(wv, bufA, v_s, v_t, vmb, kCT, kCQ, 1);

  // attention (lt tile + scores in big CDNA5 LDS)
  size_t smem = (size_t)16 * kNBP * 2 + (size_t)16 * kL * 4;  // 193,024 B
  attention_kernel<<<dim3(kL / 16, kB), 256, smem, stream>>>(qfb, kfb, vmb, lutT, rel, ctx);

  // out projection -> f32 [B, CO, H, W]
  gemm_bn_act<<<dim3(2, kCO / 16, kB), 256, 0, stream>>>(wo, ctx, o_s, o_t, d_out, kCO, kCT, 2);
}
